// PointNet_Line_22273700397308
// MI455X (gfx1250) — matrix operand order repaired and checked by reading, not compiled, not run on hardware
//
#include <hip/hip_runtime.h>

#define B_SZ  4096
#define MAXN_ 25

typedef __attribute__((ext_vector_type(16))) __bf16 v16bf;
typedef __attribute__((ext_vector_type(8)))  __bf16 v8bf;
typedef __attribute__((ext_vector_type(8)))  float  v8f;

__device__ __forceinline__ v8f zero_v8f() {
  v8f z;
#pragma unroll
  for (int i = 0; i < 8; ++i) z[i] = 0.0f;
  return z;
}

// Generic N-tile GEMM step: A (bf16, row-major, in LDS) x B^T (bf16 [N][K], global).
// Computes MSUB 16-row subtiles for one 16-wide N tile, accumulating f32.
template <int MSUB>
__device__ __forceinline__ void gemm_nt(const __bf16* aLds, int aLd,
                                        const __bf16* __restrict__ bt, int Ktot,
                                        int nbase, int ksteps,
                                        v8f* acc, int lane) {
  const int l15  = lane & 15;
  const int half = lane >> 4;
  const int kb8  = half * 8;    // A: K sub-base per half-wave
  const int kh   = half * 8;    // B: K sub-base per half-wave
  const __bf16* brow = bt + (size_t)(nbase + l15) * Ktot + kh;
  for (int ks = 0; ks < ksteps; ++ks) {
    const __bf16* bp = brow + ks * 32;
    v8bf b0 = *(const v8bf*)bp;
    v8bf b1 = *(const v8bf*)(bp + 16);
    v16bf bf;
#pragma unroll
    for (int e = 0; e < 8; ++e) { bf[e] = b0[e]; bf[e + 8] = b1[e]; }
#pragma unroll
    for (int m = 0; m < MSUB; ++m) {
      const __bf16* ap = aLds + (m * 16 + l15) * aLd + ks * 32 + kb8;
      v8bf a0 = *(const v8bf*)ap;
      v8bf a1 = *(const v8bf*)(ap + 16);
      v16bf af;
#pragma unroll
      for (int e = 0; e < 8; ++e) { af[e] = a0[e]; af[e + 8] = a1[e]; }
      acc[m] = __builtin_amdgcn_wmma_f32_16x16x32_bf16(
          false, af, false, bf, (short)0, acc[m], false, false);
    }
  }
}

// ---------------- Kernel 0: transpose + bf16-convert weights ----------------
__global__ __launch_bounds__(256) void prep_weights_kernel(
    const float* __restrict__ lw1, const float* __restrict__ lw2,
    const float* __restrict__ fw, __bf16* __restrict__ lw1t,
    __bf16* __restrict__ lw2t, __bf16* __restrict__ fwt) {
  unsigned id = blockIdx.x * 256u + threadIdx.x;
  if (id < 131072u) {                       // lw1 [256][512] -> lw1t [512][256]
    unsigned n = id >> 8, k = id & 255u;
    lw1t[n * 256 + k] = (__bf16)lw1[k * 512 + n];
  } else if (id < 655360u) {                // lw2 [512][1024] -> lw2t [1024][512]
    unsigned i = id - 131072u;
    unsigned n = i >> 9, k = i & 511u;
    lw2t[n * 512 + k] = (__bf16)lw2[k * 1024 + n];
  } else if (id < 1179648u) {               // fw [1024][511] -> fwt [512][1024], row 511 = 0
    unsigned i = id - 655360u;
    unsigned n = i >> 10, k = i & 1023u;
    fwt[n * 1024 + k] = (n < 511u) ? (__bf16)fw[k * 511 + n] : (__bf16)0.0f;
  }
}

// ---------------- Kernel 1: corner MLP (2->64 relu ->128), h as bf16 ----------------
__global__ __launch_bounds__(256) void corner_mlp_kernel(
    const float* __restrict__ fpoc, const float* __restrict__ cw1,
    const float* __restrict__ cb1, const float* __restrict__ cw2,
    const float* __restrict__ cb2, __bf16* __restrict__ hWs) {
  __shared__ float s_cw1[128];
  __shared__ float s_cb1[64];
  __shared__ float s_cw2[64 * 128];
  __shared__ float s_cb2[128];
  int t = threadIdx.x;
  if (t < 128) s_cw1[t] = cw1[t];
  if (t < 64)  s_cb1[t] = cb1[t];
  if (t < 128) s_cb2[t] = cb2[t];
  for (int i = t; i < 64 * 128; i += 256) s_cw2[i] = cw2[i];
  __syncthreads();
  int p = blockIdx.x * 256 + t;
  bool ok = p < B_SZ * MAXN_;
  int pp = ok ? p : 0;
  float x0 = fpoc[pp * 2 + 0];
  float x1 = fpoc[pp * 2 + 1];
  float h1[64];
#pragma unroll 4
  for (int j = 0; j < 64; ++j)
    h1[j] = fmaxf(x0 * s_cw1[j] + x1 * s_cw1[64 + j] + s_cb1[j], 0.0f);
  __bf16* outp = hWs + (size_t)pp * 128;
  for (int i = 0; i < 128; ++i) {
    float a = s_cb2[i];
#pragma unroll 8
    for (int j = 0; j < 64; ++j) a += h1[j] * s_cw2[j * 128 + i];
    if (ok) outp[i] = (__bf16)a;
  }
}

// ---------------- Kernel 2: pair build + line MLP + masked maxpool ----------------
#define PAIR_LD 264   // 256 + pad (bank-conflict-free fragment reads)
#define L1_LD   520   // 512 + pad
#define POOL_LD 17    // 16 + pad

__global__ __launch_bounds__(256) void line_mlp_kernel(
    const __bf16* __restrict__ hWs, const __bf16* __restrict__ lw1t,
    const __bf16* __restrict__ lw2t, const float* __restrict__ lb1,
    const float* __restrict__ lb2, const int* __restrict__ nfpc,
    __bf16* __restrict__ pooled) {
  extern __shared__ char smem[];
  __bf16* pairLds = (__bf16*)smem;                        // 128 x PAIR_LD bf16
  __bf16* l1Lds   = (__bf16*)(smem + 128 * PAIR_LD * 2);  // 128 x L1_LD bf16
  float*  poolLds = (float*)(smem + 128 * PAIR_LD * 2 + 128 * L1_LD * 2); // 8 x 128 x POOL_LD f32

  const int t    = threadIdx.x;
  const int lane = t & 31;
  const int wv   = t >> 5;
  const int l15  = lane & 15;
  const int half = lane >> 4;
  const int wgb  = blockIdx.x * 5;

  // phase 1: pair[r][0:128]=h[slot], pair[r][128:256]=h[(slot+1)%nf]; zero if invalid
  {
    int r = t >> 1, hh = t & 1;
    int b_l = r / 25, slot = r - b_l * 25;
    int b = wgb + b_l;
    bool inb = (r < 125) && (b < B_SZ);
    int bc = inb ? b : 0;
    int nf = nfpc[bc];
    bool val = inb && (slot < nf);
    int nxt = slot + 1; if (nxt >= nf) nxt = 0;
    int ss = hh ? nxt : slot;
    if (!val) ss = 0;
    const uint4* src = (const uint4*)(hWs + ((size_t)bc * MAXN_ + ss) * 128);
    uint4* dst = (uint4*)(pairLds + r * PAIR_LD + hh * 128);
    uint4 z; z.x = z.y = z.z = z.w = 0u;
#pragma unroll
    for (int i = 0; i < 16; ++i) {
      uint4 v = src[i];
      dst[i] = val ? v : z;
    }
  }
  __syncthreads();

  // phase 2: GEMM1 [128x256]@[256x512] + lb1, relu -> l1 (bf16 in LDS)
  for (int nt = 0; nt < 4; ++nt) {
    int nbase = wv * 64 + nt * 16;
    v8f acc[8];
#pragma unroll
    for (int m = 0; m < 8; ++m) acc[m] = zero_v8f();
    gemm_nt<8>(pairLds, PAIR_LD, lw1t, 256, nbase, 8, acc, lane);
    float bv = lb1[nbase + l15];
#pragma unroll
    for (int m = 0; m < 8; ++m)
#pragma unroll
      for (int r = 0; r < 8; ++r) {
        int row = m * 16 + r + half * 8;
        l1Lds[row * L1_LD + nbase + l15] = (__bf16)fmaxf(acc[m][r] + bv, 0.0f);
      }
  }
  __syncthreads();

  // phase 3: GEMM2 [128x512]@[512x1024] + lb2, then per-b masked max over slots
  float* pb = poolLds + wv * (128 * POOL_LD);
  for (int nt = 0; nt < 8; ++nt) {
    int nbase = wv * 128 + nt * 16;
    v8f acc[8];
#pragma unroll
    for (int m = 0; m < 8; ++m) acc[m] = zero_v8f();
    gemm_nt<8>(l1Lds, L1_LD, lw2t, 512, nbase, 16, acc, lane);
    float bv = lb2[nbase + l15];
#pragma unroll
    for (int m = 0; m < 8; ++m)
#pragma unroll
      for (int r = 0; r < 8; ++r) {
        int row = m * 16 + r + half * 8;
        pb[row * POOL_LD + l15] = acc[m][r] + bv;
      }
    // wave-local pooling: slot0 always valid (nfpc>=3), so max over slots<nf
    for (int tsk = lane; tsk < 80; tsk += 32) {
      int b_l = tsk >> 4, c = tsk & 15;
      int b = wgb + b_l;
      if (b < B_SZ) {
        int nf = nfpc[b];
        float mx = pb[(b_l * 25) * POOL_LD + c];
        for (int s = 1; s < nf; ++s)
          mx = fmaxf(mx, pb[(b_l * 25 + s) * POOL_LD + c]);
        pooled[(size_t)b * 1024 + nbase + c] = (__bf16)mx;
      }
    }
  }
}

// ---------------- Kernel 3: final GEMM [4096x1024]@[1024x511] + fb ----------------
#define A3_LD 1032
__global__ __launch_bounds__(256) void final_gemm_kernel(
    const __bf16* __restrict__ pooled, const __bf16* __restrict__ fwt,
    const float* __restrict__ fb, float* __restrict__ out) {
  extern __shared__ char smem[];
  __bf16* aLds = (__bf16*)smem;  // 128 x A3_LD bf16
  const int t    = threadIdx.x;
  const int lane = t & 31;
  const int wv   = t >> 5;
  const int l15  = lane & 15;
  const int half = lane >> 4;
  const int mrow0 = blockIdx.x * 128;
  {
    int r = t >> 1, hh = t & 1;
    const uint4* src = (const uint4*)(pooled + (size_t)(mrow0 + r) * 1024 + hh * 512);
    uint4* dst = (uint4*)(aLds + r * A3_LD + hh * 512);
#pragma unroll 8
    for (int i = 0; i < 64; ++i) dst[i] = src[i];
  }
  __syncthreads();
  for (int nt = 0; nt < 4; ++nt) {
    int nbase = wv * 64 + nt * 16;
    v8f acc[8];
#pragma unroll
    for (int m = 0; m < 8; ++m) acc[m] = zero_v8f();
    gemm_nt<8>(aLds, A3_LD, fwt, 1024, nbase, 32, acc, lane);
    int col = nbase + l15;
    if (col < 511) {
      float bv = fb[col];
#pragma unroll
      for (int m = 0; m < 8; ++m)
#pragma unroll
        for (int r = 0; r < 8; ++r) {
          int row = mrow0 + m * 16 + r + half * 8;
          out[(size_t)row * 511 + col] = acc[m][r] + bv;
        }
    }
  }
}

extern "C" void kernel_launch(void* const* d_in, const int* in_sizes, int n_in,
                              void* d_out, int out_size, void* d_ws, size_t ws_size,
                              hipStream_t stream) {
  const float* fpoc = (const float*)d_in[0];
  const float* cw1  = (const float*)d_in[1];
  const float* cb1  = (const float*)d_in[2];
  const float* cw2  = (const float*)d_in[3];
  const float* cb2  = (const float*)d_in[4];
  const float* lw1  = (const float*)d_in[5];
  const float* lb1  = (const float*)d_in[6];
  const float* lw2  = (const float*)d_in[7];
  const float* lb2  = (const float*)d_in[8];
  const float* fw   = (const float*)d_in[9];
  const float* fb   = (const float*)d_in[10];
  const int*   nfpc = (const int*)d_in[11];

  char* ws = (char*)d_ws;
  __bf16* lw1t   = (__bf16*)(ws);             //   512*256*2 = 262144
  __bf16* lw2t   = (__bf16*)(ws + 262144);    //  1024*512*2 = 1048576
  __bf16* fwt    = (__bf16*)(ws + 1310720);   //   512*1024*2 = 1048576
  __bf16* hWs    = (__bf16*)(ws + 2359296);   // 4096*25*128*2 = 26214400
  __bf16* pooled = (__bf16*)(ws + 28573696);  // 4096*1024*2 = 8388608  (total ~36.9 MB)
  float* out = (float*)d_out;

  prep_weights_kernel<<<4608, 256, 0, stream>>>(lw1, lw2, fw, lw1t, lw2t, fwt);
  corner_mlp_kernel<<<400, 256, 0, stream>>>(fpoc, cw1, cb1, cw2, cb2, hWs);

  size_t smem2 = 128 * PAIR_LD * 2 + 128 * L1_LD * 2 + 8 * 128 * POOL_LD * 4; // 270336
  line_mlp_kernel<<<820, 256, smem2, stream>>>(hWs, lw1t, lw2t, lb1, lb2, nfpc, pooled);

  size_t smem3 = 128 * A3_LD * 2; // 264192
  final_gemm_kernel<<<32, 256, smem3, stream>>>(pooled, fwt, fb, out);
}
